// RNNSequenceEncoder_56547539419605
// MI455X (gfx1250) — compile-verified, matrix operand order checked by reading, not run on hardware
//
#include <hip/hip_runtime.h>
#include <hip/hip_bf16.h>
#include <math.h>

// ---------------------------------------------------------------------------
// LSTM stack for MI455X (gfx1250): bf16 WMMA (f32 accumulate) everywhere.
//   B=64, T=512, D=H=1024, L=2, gates packed [i,f,g,o] (4H = 4096).
// ---------------------------------------------------------------------------

typedef __bf16 bf16_t;
typedef __attribute__((ext_vector_type(16))) __bf16 v16bf;
typedef __attribute__((ext_vector_type(8)))  __bf16 v8bf;
typedef __attribute__((ext_vector_type(8)))  float   v8f;

#define BB    64
#define TT    512
#define DD    1024
#define HH    1024
#define G4H   4096
#define MROWS (BB * TT)   // 32768

// ---- WMMA fragment helpers -------------------------------------------------
// 16-bit A 16x32 layout (ISA 7.12.2): lane L holds row (L%16); K elements:
//   e in [0,8)  -> K = (L/16)*8 + e          (VGPR0..3)
//   e in [8,16) -> K = 16 + (L/16)*8 + (e-8) (VGPR4..7)
// i.e. two contiguous 8-element (16-byte) chunks at +0 and +16 halfwords.
// B (32x16) stored as B^T row-major ([N][K]) uses the identical pattern with
// lane holding column (L%16).
__device__ __forceinline__ v16bf frag_ld(const bf16_t* p) {
  union U { struct { v8bf lo, hi; } s; v16bf v; } u;
  u.s.lo = *(const v8bf*)(p);
  u.s.hi = *(const v8bf*)(p + 16);
  return u.v;
}

__device__ __forceinline__ v8f wmma_bf16(v16bf a, v16bf b, v8f c) {
  // (neg_a, A, neg_b, B, c_mod, C, reuse_a, reuse_b)
  return __builtin_amdgcn_wmma_f32_16x16x32_bf16(false, a, false, b, (short)0,
                                                 c, false, false);
}

__device__ __forceinline__ float sigmoidf_(float x) {
  return 1.0f / (1.0f + __expf(-x));
}

// ---- grid-wide barrier (persistent cooperative kernel) ---------------------
__device__ __forceinline__ void grid_sync(unsigned* cnt, unsigned* gen,
                                          unsigned nblocks) {
  __syncthreads();
  if (threadIdx.x == 0) {
    __threadfence();  // release our global stores
    unsigned g = __hip_atomic_load(gen, __ATOMIC_RELAXED,
                                   __HIP_MEMORY_SCOPE_AGENT);
    unsigned a = __hip_atomic_fetch_add(cnt, 1u, __ATOMIC_ACQ_REL,
                                        __HIP_MEMORY_SCOPE_AGENT);
    if (a == nblocks - 1u) {
      __hip_atomic_store(cnt, 0u, __ATOMIC_RELAXED, __HIP_MEMORY_SCOPE_AGENT);
      __hip_atomic_fetch_add(gen, 1u, __ATOMIC_RELEASE,
                             __HIP_MEMORY_SCOPE_AGENT);
    } else {
      while (__hip_atomic_load(gen, __ATOMIC_ACQUIRE,
                               __HIP_MEMORY_SCOPE_AGENT) == g)
        __builtin_amdgcn_s_sleep(1);
    }
    __threadfence();  // acquire: subsequent vector loads see fresh data
  }
  __syncthreads();
}

// ---- one-time conversion kernels ------------------------------------------
// x f32 [B,T,D] -> bf16 time-major [T*B, D]
__global__ void convert_input_tmajor(const float* __restrict__ x,
                                     bf16_t* __restrict__ xo) {
  size_t e = (size_t)blockIdx.x * blockDim.x + threadIdx.x;
  if (e >= (size_t)MROWS * DD) return;
  size_t d  = e % DD;
  size_t bt = e / DD;
  size_t t  = bt % TT;
  size_t b  = bt / TT;
  xo[(t * BB + b) * DD + d] = (bf16_t)x[e];
}

// W f32 [K,N] -> WT bf16 [N,K]; LDS-tiled so both global sides are coalesced.
// grid = (N/32, K/32), block = 256 (32 x 8).
#define TRT 32
__global__ __launch_bounds__(256) void transpose_to_bf16(
    const float* __restrict__ W, bf16_t* __restrict__ WT, int K, int N) {
  __shared__ bf16_t tile[TRT][TRT + 2];  // +2 halfwords: kill bank conflicts
  const int n0 = blockIdx.x * TRT;
  const int k0 = blockIdx.y * TRT;
  const int tx = threadIdx.x & 31;
  const int ty = threadIdx.x >> 5;  // 0..7
#pragma unroll
  for (int i = 0; i < TRT; i += 8) {  // coalesced read of W rows
    tile[ty + i][tx] = (bf16_t)W[(size_t)(k0 + ty + i) * N + (n0 + tx)];
  }
  __syncthreads();
#pragma unroll
  for (int i = 0; i < TRT; i += 8) {  // coalesced write of WT rows
    WT[(size_t)(n0 + ty + i) * K + (k0 + tx)] = tile[tx][ty + i];
  }
}

// ---- batched GEMM: out = A[M,K] x B[K,N] (+bias), B given transposed ------
// Per-wave macro tile: 32(M) x 64(N) = 8 accumulators; 8 WMMAs per K-step.
// mode 0: store f32 row-major [M,N]          (xg = x@Wx + b)
// mode 1: store bf16 row-major [M,N]         (layer-0 projection -> next input)
// mode 2: store f32 remapped [T,B]->[B,T,D]  (final projection -> d_out)
__global__ __launch_bounds__(256) void gemm_wmma(
    const bf16_t* __restrict__ A, const bf16_t* __restrict__ Bt,
    const float* __restrict__ bias, float* __restrict__ outF,
    bf16_t* __restrict__ outBf, int M, int N, int K, int mode) {
  const int wib = threadIdx.x >> 5;
  const int gid = blockIdx.x * (blockDim.x >> 5) + wib;
  const int Mt = M >> 5, Nt = N >> 6;
  if (gid >= Mt * Nt) return;
  const int mi = gid % Mt, ni = gid / Mt;
  const int m0 = mi << 5, n0 = ni << 6;
  const int lane = threadIdx.x & 31;
  const int r16 = lane & 15, half = lane >> 4;

  v8f acc[2][4] = {};
  const bf16_t* a0p = A + (size_t)(m0 + r16) * K + half * 8;
  const bf16_t* a1p = a0p + (size_t)16 * K;
  const bf16_t* bp0 = Bt + (size_t)(n0 + r16) * K + half * 8;

#pragma unroll 2
  for (int k0 = 0; k0 < K; k0 += 32) {
    v16bf a0 = frag_ld(a0p + k0);
    v16bf a1 = frag_ld(a1p + k0);
    v16bf b0 = frag_ld(bp0 + k0);
    v16bf b1 = frag_ld(bp0 + (size_t)16 * K + k0);
    v16bf b2 = frag_ld(bp0 + (size_t)32 * K + k0);
    v16bf b3 = frag_ld(bp0 + (size_t)48 * K + k0);
    acc[0][0] = wmma_bf16(a0, b0, acc[0][0]);
    acc[0][1] = wmma_bf16(a0, b1, acc[0][1]);
    acc[0][2] = wmma_bf16(a0, b2, acc[0][2]);
    acc[0][3] = wmma_bf16(a0, b3, acc[0][3]);
    acc[1][0] = wmma_bf16(a1, b0, acc[1][0]);
    acc[1][1] = wmma_bf16(a1, b1, acc[1][1]);
    acc[1][2] = wmma_bf16(a1, b2, acc[1][2]);
    acc[1][3] = wmma_bf16(a1, b3, acc[1][3]);
  }

#pragma unroll
  for (int i = 0; i < 2; ++i) {
#pragma unroll
    for (int j = 0; j < 4; ++j) {
      const int col = n0 + j * 16 + r16;
      const float bv = bias ? bias[col] : 0.0f;
#pragma unroll
      for (int r = 0; r < 8; ++r) {
        const int row = m0 + i * 16 + r + half * 8;
        const float v = acc[i][j][r] + bv;
        if (mode == 0) {
          outF[(size_t)row * N + col] = v;
        } else if (mode == 1) {
          outBf[(size_t)row * N + col] = (bf16_t)v;
        } else {  // mode 2: row = t*BB + b  ->  out[b][t][col], D == N
          const int b = row & (BB - 1);
          const int t = row >> 6;
          outF[((size_t)b * TT + t) * (size_t)N + col] = v;
        }
      }
    }
  }
}

// ---- persistent LSTM recurrence -------------------------------------------
// grid = 64 blocks x 128 threads = 256 waves; wave w owns batch tile
// m0 = (w>>6)*16 and hidden columns c0 = (w&63)*16 across ALL FOUR gate
// blocks (i,f,g,o), so the entire cell update stays in registers.
// h is ping-pong double-buffered (bf16) -> one grid barrier per step.
// Wh^T (8.4 MB bf16) stays resident in the 192 MB L2 across all 512 steps.
__global__ __launch_bounds__(128) void lstm_scan(
    const float* __restrict__ xg,    // [T*B, 4096] time-major (x@Wx + b)
    const bf16_t* __restrict__ WhT,  // [4096, 1024]
    bf16_t* __restrict__ hbuf,       // [2][64*1024] ping-pong
    float* __restrict__ c_state,     // [64*1024]
    bf16_t* __restrict__ h_all,      // [T*B, 1024]
    unsigned* __restrict__ cnt, unsigned* __restrict__ gen) {
  // zero-init c and the t=0 read buffer
  for (size_t e = (size_t)blockIdx.x * blockDim.x + threadIdx.x;
       e < (size_t)BB * HH; e += (size_t)gridDim.x * blockDim.x) {
    c_state[e] = 0.0f;
    hbuf[e] = (bf16_t)0.0f;
  }
  grid_sync(cnt, gen, gridDim.x);

  const int wid = threadIdx.x >> 5;
  const int gw = blockIdx.x * (blockDim.x >> 5) + wid;  // 0..255
  const int m0 = (gw >> 6) << 4;                        // batch tile row
  const int c0 = (gw & 63) << 4;                        // hidden col tile
  const int lane = threadIdx.x & 31;
  const int r16 = lane & 15, half = lane >> 4;

  const bf16_t* bIp = WhT + (size_t)(0 * HH + c0 + r16) * HH + half * 8;
  const bf16_t* bFp = bIp + (size_t)1 * HH * HH;
  const bf16_t* bGp = bIp + (size_t)2 * HH * HH;
  const bf16_t* bOp = bIp + (size_t)3 * HH * HH;

  for (int t = 0; t < TT; ++t) {
    const bf16_t* hr = hbuf + (size_t)(t & 1) * (BB * HH);
    bf16_t* hw = hbuf + (size_t)((t + 1) & 1) * (BB * HH);

    v8f aI = {}, aF = {}, aG = {}, aO = {};
    const bf16_t* ap = hr + (size_t)(m0 + r16) * HH + half * 8;
#pragma unroll 2
    for (int k0 = 0; k0 < HH; k0 += 32) {
      v16bf a = frag_ld(ap + k0);
      aI = wmma_bf16(a, frag_ld(bIp + k0), aI);
      aF = wmma_bf16(a, frag_ld(bFp + k0), aF);
      aG = wmma_bf16(a, frag_ld(bGp + k0), aG);
      aO = wmma_bf16(a, frag_ld(bOp + k0), aO);
    }

    const size_t xbase = (size_t)t * BB * G4H;
#pragma unroll
    for (int r = 0; r < 8; ++r) {
      const int brow = m0 + r + half * 8;
      const int col = c0 + r16;
      const size_t xi = xbase + (size_t)brow * G4H + col;
      const float gi = aI[r] + xg[xi];
      const float gf = aF[r] + xg[xi + 1024];
      const float gg = aG[r] + xg[xi + 2048];
      const float go = aO[r] + xg[xi + 3072];
      const size_t ci = (size_t)brow * HH + col;
      const float c = sigmoidf_(gf) * c_state[ci] + sigmoidf_(gi) * tanhf(gg);
      const float h = sigmoidf_(go) * tanhf(c);
      c_state[ci] = c;
      const bf16_t hb = (bf16_t)h;
      hw[ci] = hb;
      h_all[((size_t)t * BB + brow) * HH + col] = hb;
    }
    grid_sync(cnt, gen, gridDim.x);
  }
}

// ---------------------------------------------------------------------------
extern "C" void kernel_launch(void* const* d_in, const int* in_sizes, int n_in,
                              void* d_out, int out_size, void* d_ws,
                              size_t ws_size, hipStream_t stream) {
  (void)in_sizes; (void)n_in; (void)out_size;
  const float* x  = (const float*)d_in[0];
  const float* Wx = (const float*)d_in[1];  // [2,1024,4096]
  const float* Wh = (const float*)d_in[2];  // [2,1024,4096]
  const float* bg = (const float*)d_in[3];  // [2,4096]
  const float* Wp = (const float*)d_in[4];  // [2,1024,1024]
  const float* bp = (const float*)d_in[5];  // [2,1024]
  float* out = (float*)d_out;

  char* ws = (char*)d_ws;
  size_t off = 0;
  auto take = [&](size_t bytes) -> char* {
    char* p = ws + off;
    off += (bytes + 255) & ~(size_t)255;
    return p;
  };
  unsigned* ctrl = (unsigned*)take(256);                       // cnt, gen
  bf16_t* x_cur = (bf16_t*)take((size_t)MROWS * DD * 2);       // layer input
  bf16_t* h_all = (bf16_t*)take((size_t)MROWS * HH * 2);       // hidden seq
  float*  xg    = (float*) take((size_t)MROWS * G4H * 4);      // gate pre-acts
  bf16_t* WxT   = (bf16_t*)take((size_t)2 * G4H * DD * 2);
  bf16_t* WhT   = (bf16_t*)take((size_t)2 * G4H * HH * 2);
  bf16_t* WpT   = (bf16_t*)take((size_t)2 * DD * HH * 2);
  bf16_t* hbuf  = (bf16_t*)take((size_t)2 * BB * HH * 2);
  float*  cst   = (float*) take((size_t)BB * HH * 4);
  if (off > ws_size) return;  // workspace too small: bail deterministically

  (void)hipMemsetAsync(ctrl, 0, 256, stream);  // barrier counters

  {  // input conversion to bf16 time-major
    size_t n = (size_t)MROWS * DD;
    convert_input_tmajor<<<dim3((unsigned)((n + 255) / 256)), dim3(256), 0,
                           stream>>>(x, x_cur);
  }
  for (int l = 0; l < 2; ++l) {  // weight transpose+convert (LDS-tiled)
    {
      size_t n = (size_t)DD * G4H;
      transpose_to_bf16<<<dim3(G4H / TRT, DD / TRT), dim3(256), 0, stream>>>(
          Wx + (size_t)l * n, WxT + (size_t)l * n, DD, G4H);
      transpose_to_bf16<<<dim3(G4H / TRT, HH / TRT), dim3(256), 0, stream>>>(
          Wh + (size_t)l * n, WhT + (size_t)l * n, HH, G4H);
    }
    {
      size_t n = (size_t)HH * DD;
      transpose_to_bf16<<<dim3(DD / TRT, HH / TRT), dim3(256), 0, stream>>>(
          Wp + (size_t)l * n, WpT + (size_t)l * n, HH, DD);
    }
  }

  for (int l = 0; l < 2; ++l) {
    const bf16_t* WxTl = WxT + (size_t)l * G4H * DD;
    const bf16_t* WhTl = WhT + (size_t)l * G4H * HH;
    const bf16_t* WpTl = WpT + (size_t)l * DD * HH;
    const float* bl  = bg + (size_t)l * G4H;
    const float* bpl = bp + (size_t)l * DD;

    {  // xg = x_cur @ Wx + b   (M=32768, N=4096, K=1024)
      int waves = (MROWS / 32) * (G4H / 64);  // 65536
      gemm_wmma<<<dim3(waves / 8), dim3(256), 0, stream>>>(
          x_cur, WxTl, bl, xg, nullptr, MROWS, G4H, DD, 0);
    }
    // sequential recurrence, persistent cooperative kernel
    lstm_scan<<<dim3(64), dim3(128), 0, stream>>>(xg, WhTl, hbuf, cst, h_all,
                                                  ctrl, ctrl + 1);
    {  // projection (M=32768, N=1024, K=1024)
      int waves = (MROWS / 32) * (HH / 64);  // 16384
      int mode = (l == 0) ? 1 : 2;
      gemm_wmma<<<dim3(waves / 8), dim3(256), 0, stream>>>(
          h_all, WpTl, bpl, (mode == 2) ? out : nullptr,
          (mode == 1) ? x_cur : nullptr, MROWS, HH, HH, mode);
    }
  }
}
